// Net_81432579932954
// MI455X (gfx1250) — matrix operand order repaired and checked by reading, not compiled, not run on hardware
//
#include <hip/hip_runtime.h>
#include <hip/hip_bf16.h>

typedef __attribute__((ext_vector_type(16))) __bf16       v16bf;
typedef __attribute__((ext_vector_type(8)))  float        v8f;
typedef __attribute__((ext_vector_type(4)))  unsigned int u32x4;
typedef __attribute__((ext_vector_type(8)))  int          i32x8;
typedef __attribute__((ext_vector_type(4)))  int          i32x4;

#define NN   8192   // nodes
#define EE   16384  // edges
#define GG   512    // graphs
#define INF  235    // input feature dim
#define HH   64     // hidden dim
#define EDF  52     // edge feature dim
#define KP   256    // padded IN (multiple of 32 for WMMA K)
#define JD   8192   // U columns = 128 (k) * 64 (h)

// ---- Tensor Data Mover availability / toolchain arity detection ---------
#if defined(__HIP_DEVICE_COMPILE__) && defined(__gfx1250__) && \
    __has_builtin(__builtin_amdgcn_tensor_load_to_lds)
#define HAVE_TDM 1
#else
#define HAVE_TDM 0
#endif

#if __has_include(<hip/amd_detail/amd_gfx1250_TDM.h>)
#define TDM_6ARGS 1   // amdgpu-toolchain (clang-23 / therock): 6-arg builtin
#else
#define TDM_6ARGS 0   // ROCm 7.2 (clang-22): 5-arg builtin
#endif

// Issue a 2D TDM tile load: `rows` rows of `rowlen` elements (2B each),
// row stride `stride` elements, global -> LDS offset `lds_off`.
__device__ __forceinline__ void tdm_load_2d(unsigned lds_off, const void* gaddr,
                                            unsigned rows, unsigned rowlen,
                                            unsigned stride) {
#if HAVE_TDM
    unsigned long long ga = (unsigned long long)gaddr;
    u32x4 g0;
    g0[0] = 1u;                                         // count=1, user mode
    g0[1] = lds_off;                                    // lds_addr[31:0]
    g0[2] = (unsigned)(ga & 0xFFFFFFFFu);               // global_addr[31:0]
    g0[3] = (unsigned)((ga >> 32) & 0x1FFFFFFu)         // global_addr[56:32]
          | (2u << 30);                                 // type = 2 ("image")
    i32x8 g1;
    g1[0] = (int)(1u << 16);                            // data_size=1 (2 bytes)
    g1[1] = (int)((rowlen & 0xFFFFu) << 16);            // tensor_dim0[15:0]
    g1[2] = (int)(((rowlen >> 16) & 0xFFFFu)            // tensor_dim0[31:16]
          | ((rows & 0xFFFFu) << 16));                  // tensor_dim1[15:0]
    g1[3] = (int)(((rows >> 16) & 0xFFFFu)              // tensor_dim1[31:16]
          | ((rowlen & 0xFFFFu) << 16));                // tile_dim0
    g1[4] = (int)(rows & 0xFFFFu);                      // tile_dim1 (tile_dim2=0)
    g1[5] = (int)stride;                                // tensor_dim0_stride[31:0]
    g1[6] = 0;                                          // stride hi / dim1_stride lo
    g1[7] = 0;                                          // dim1_stride hi
    i32x4 z4 = {0, 0, 0, 0};
#if TDM_6ARGS
    i32x8 z8 = {0, 0, 0, 0, 0, 0, 0, 0};
    __builtin_amdgcn_tensor_load_to_lds(g0, g1, z4, z4, z8, 0);
#else
    __builtin_amdgcn_tensor_load_to_lds(g0, g1, z4, z4, 0);
#endif
#else
    (void)lds_off; (void)gaddr; (void)rows; (void)rowlen; (void)stride;
#endif
}

// ---------------- edge MLP layer 1: he[E,128] = edge_attr @ W1^T + b1 ----
__global__ void __launch_bounds__(256)
edge_mlp1(const float* __restrict__ ea, const float* __restrict__ W1,
          const float* __restrict__ b1, float* __restrict__ he) {
    __shared__ float sa[2][EDF];
    int t = threadIdx.x;
    int e0 = blockIdx.x * 2;
    if (t < 2 * EDF) sa[t / EDF][t % EDF] = ea[(size_t)e0 * EDF + t];
    __syncthreads();
    int le = t >> 7;       // local edge 0/1
    int o  = t & 127;      // output channel
    float acc = b1[o];
    const float* w = W1 + (size_t)o * EDF;
#pragma unroll 4
    for (int c = 0; c < EDF; ++c) acc += sa[le][c] * w[c];
    he[(size_t)(e0 + le) * 128 + o] = acc;
}

// ---------------- pack x -> bf16 [N, KP] (zero-padded K) -----------------
__global__ void __launch_bounds__(256)
pack_x(const float* __restrict__ x, __bf16* __restrict__ Xp) {
    int n = blockIdx.x, i = threadIdx.x;
    float v = (i < INF) ? x[(size_t)n * INF + i] : 0.0f;
    Xp[(size_t)n * KP + i] = (__bf16)v;
}

// ---- pack W2 -> BmT bf16 [JD, KP]; BmT[k*64+h][i] = W2[(i*64+h)*128+k] --
__global__ void __launch_bounds__(256)
pack_w2(const float* __restrict__ W2, __bf16* __restrict__ BmT) {
    int j = blockIdx.x, i = threadIdx.x;
    int h = j & 63, k = j >> 6;
    float v = (i < INF) ? W2[((size_t)i * HH + h) * 128 + k] : 0.0f;
    BmT[(size_t)j * KP + i] = (__bf16)v;
}

// -------- per-node: Bbias = x@b2_mat ; xroot = x@root + conv_b ----------
__global__ void __launch_bounds__(256)
node_linear(const float* __restrict__ x, const float* __restrict__ b2,
            const float* __restrict__ root, const float* __restrict__ conv_b,
            float* __restrict__ Bbias, float* __restrict__ xroot) {
    __shared__ float sx[4][INF];
    int t = threadIdx.x;
    int n0 = blockIdx.x * 4;
    for (int idx = t; idx < 4 * INF; idx += 256)
        sx[idx / INF][idx % INF] = x[(size_t)n0 * INF + idx];
    __syncthreads();
    int le = t >> 6, h = t & 63;
    int n = n0 + le;
    float ab = 0.0f, ar = 0.0f;
    for (int i = 0; i < INF; ++i) {
        float xv = sx[le][i];
        ab += xv * b2[(size_t)i * HH + h];
        ar += xv * root[(size_t)i * HH + h];
    }
    Bbias[(size_t)n * HH + h] = ab;
    xroot[(size_t)n * HH + h] = ar + conv_b[h];
}

// --------- big GEMM: U[NN, JD] (bf16) = Xp[NN,KP] @ BmT[JD,KP]^T --------
// WMMA bf16 16x16x32, wave32, 8 waves/block, block tile M=128 x N=64.
// The block's B panel (64 rows x 256 K = 32 KB) is staged into LDS once by
// the Tensor Data Mover (wave 0 issues TENSOR_LOAD_TO_LDS, waits TENSORcnt,
// then the workgroup barrier publishes it); all 8 waves read B from LDS.
__global__ void __launch_bounds__(256)
gemm_wmma(const __bf16* __restrict__ A, const __bf16* __restrict__ B,
          __bf16* __restrict__ U) {
    __shared__ __bf16 sB[64 * KP];   // 32 KB B panel: rows j0..j0+63
    const int t    = threadIdx.x;
    const int lane = t & 31;
    const int wave = t >> 5;
    const int l15  = lane & 15;
    const int lh   = lane >> 4;      // which K-half this lane holds
    const int m0   = blockIdx.y * 128 + wave * 16;
    const int j0   = blockIdx.x * 64;

#if HAVE_TDM
    if (wave == 0) {
        tdm_load_2d((unsigned)(size_t)sB, B + (size_t)j0 * KP,
                    /*rows=*/64, /*rowlen=*/KP, /*stride=*/KP);
        __builtin_amdgcn_s_wait_tensorcnt(0);
    }
#else
    for (int idx = t * 8; idx < 64 * KP; idx += 256 * 8)
        *(float4*)&sB[idx] = *(const float4*)(B + (size_t)j0 * KP + idx);
#endif
    __syncthreads();

    v8f acc0 = {}, acc1 = {}, acc2 = {}, acc3 = {};
    const __bf16* arow = A + (size_t)(m0 + l15) * KP;

    for (int k0 = 0; k0 < KP; k0 += 32) {
        union Frag { v16bf v; float4 f4[2]; } a, b0, b1, b2, b3;
        // A 16x32 bf16 fragment: lane holds 16 consecutive K[lh*16..+15]
        a.f4[0] = *(const float4*)(arow + k0 + lh * 16);
        a.f4[1] = *(const float4*)(arow + k0 + lh * 16 + 8);
        // B 32x16 bf16 fragments from LDS: lane holds 16 consecutive K
        const int bof = l15 * KP + k0 + lh * 16;
        b0.f4[0] = *(const float4*)(&sB[bof]);
        b0.f4[1] = *(const float4*)(&sB[bof + 8]);
        b1.f4[0] = *(const float4*)(&sB[bof + 16 * KP]);
        b1.f4[1] = *(const float4*)(&sB[bof + 16 * KP + 8]);
        b2.f4[0] = *(const float4*)(&sB[bof + 32 * KP]);
        b2.f4[1] = *(const float4*)(&sB[bof + 32 * KP + 8]);
        b3.f4[0] = *(const float4*)(&sB[bof + 48 * KP]);
        b3.f4[1] = *(const float4*)(&sB[bof + 48 * KP + 8]);

        acc0 = __builtin_amdgcn_wmma_f32_16x16x32_bf16(false, a.v, false, b0.v,
                                                       (short)0, acc0, false, false);
        acc1 = __builtin_amdgcn_wmma_f32_16x16x32_bf16(false, a.v, false, b1.v,
                                                       (short)0, acc1, false, false);
        acc2 = __builtin_amdgcn_wmma_f32_16x16x32_bf16(false, a.v, false, b2.v,
                                                       (short)0, acc2, false, false);
        acc3 = __builtin_amdgcn_wmma_f32_16x16x32_bf16(false, a.v, false, b3.v,
                                                       (short)0, acc3, false, false);
    }

    // D layout: lane -> column j0 + l15; VGPR r -> row m0 + r + lh*8
#pragma unroll
    for (int r = 0; r < 8; ++r) {
        size_t m = (size_t)(m0 + r + lh * 8);
        __bf16* up = U + m * JD + j0 + l15;
        up[0]  = (__bf16)acc0[r];
        up[16] = (__bf16)acc1[r];
        up[32] = (__bf16)acc2[r];
        up[48] = (__bf16)acc3[r];
    }
}

// ------ per-edge matvec + scatter: agg[dst] += he[e] . U[src] + Bbias ----
__global__ void __launch_bounds__(256)
edge_msg(const __bf16* __restrict__ U, const float* __restrict__ he,
         const float* __restrict__ Bbias, const int* __restrict__ ei,
         float* __restrict__ agg) {
    __shared__ float sh[4][128];
    int t = threadIdx.x;
    int e0 = blockIdx.x * 4;
    for (int idx = t; idx < 4 * 128; idx += 256)
        sh[idx >> 7][idx & 127] = he[(size_t)e0 * 128 + idx];
    __syncthreads();
    int le = t >> 6, h = t & 63;
    int e = e0 + le;
    int s = ei[e];        // src
    int d = ei[EE + e];   // dst
    const __bf16* urow = U + (size_t)s * JD + h;   // coalesced over h
    float acc = Bbias[(size_t)s * HH + h];
#pragma unroll 8
    for (int k = 0; k < 128; ++k)
        acc += sh[le][k] * (float)urow[(size_t)k * 64];
    atomicAdd(&agg[(size_t)d * HH + h], acc);
}

// ---------- out = relu(agg + xroot); pool sums per graph -----------------
__global__ void __launch_bounds__(256)
relu_pool(const float* __restrict__ agg, const float* __restrict__ xroot,
          const int* __restrict__ batch, float* __restrict__ psum,
          float* __restrict__ cnt) {
    int idx = blockIdx.x * 256 + threadIdx.x;   // n*64 + h
    int n = idx >> 6, h = idx & 63;
    float v = agg[idx] + xroot[idx];
    v = v > 0.0f ? v : 0.0f;
    int g = batch[n];
    atomicAdd(&psum[(size_t)g * HH + h], v);
    if (h == 0) atomicAdd(&cnt[g], 1.0f);
}

// ---------------- dense MLP head, one block per graph --------------------
__global__ void __launch_bounds__(256)
head_mlp(const float* __restrict__ psum, const float* __restrict__ cnt,
         const float* __restrict__ fw1, const float* __restrict__ fb1,
         const float* __restrict__ fw2, const float* __restrict__ fb2,
         const float* __restrict__ fw3, const float* __restrict__ fb3,
         const float* __restrict__ fw4, const float* __restrict__ fb4,
         float* __restrict__ out) {
    __shared__ float sp[64], s1[128], s2[256], s3[128];
    int g = blockIdx.x, t = threadIdx.x;
    float c = cnt[g]; c = c > 1.0f ? c : 1.0f;
    if (t < 64) sp[t] = psum[(size_t)g * 64 + t] / c;
    __syncthreads();
    if (t < 128) {
        float a = fb1[t];
        for (int k = 0; k < 64; ++k) a += fw1[(size_t)t * 64 + k] * sp[k];
        s1[t] = a > 0.0f ? a : 0.0f;
    }
    __syncthreads();
    {
        float a = fb2[t];
        for (int k = 0; k < 128; ++k) a += fw2[(size_t)t * 128 + k] * s1[k];
        s2[t] = a > 0.0f ? a : 0.0f;
    }
    __syncthreads();
    if (t < 128) {
        float a = fb3[t];
        for (int k = 0; k < 256; ++k) a += fw3[(size_t)t * 256 + k] * s2[k];
        s3[t] = a > 0.0f ? a : 0.0f;
    }
    __syncthreads();
    if (t == 0) {
        float a = fb4[0];
        for (int k = 0; k < 128; ++k) a += fw4[k] * s3[k];
        out[g] = a;
    }
}

extern "C" void kernel_launch(void* const* d_in, const int* in_sizes, int n_in,
                              void* d_out, int out_size, void* d_ws, size_t ws_size,
                              hipStream_t stream) {
    const float* x      = (const float*)d_in[0];
    const int*   ei     = (const int*)  d_in[1];
    const float* eattr  = (const float*)d_in[2];
    const int*   batch  = (const int*)  d_in[3];
    const float* W1     = (const float*)d_in[4];
    const float* b1     = (const float*)d_in[5];
    const float* W2     = (const float*)d_in[6];
    const float* b2     = (const float*)d_in[7];
    const float* root   = (const float*)d_in[8];
    const float* conv_b = (const float*)d_in[9];
    const float* fw1 = (const float*)d_in[10]; const float* fb1 = (const float*)d_in[11];
    const float* fw2 = (const float*)d_in[12]; const float* fb2 = (const float*)d_in[13];
    const float* fw3 = (const float*)d_in[14]; const float* fb3 = (const float*)d_in[15];
    const float* fw4 = (const float*)d_in[16]; const float* fb4 = (const float*)d_in[17];
    float* out = (float*)d_out;

    char* ws = (char*)d_ws;
    // workspace layout (bytes)
    __bf16* U     = (__bf16*)(ws + 0);                 // 8192*8192*2   = 134217728
    float*  he    = (float*) (ws + 134217728);         // 16384*128*4   =   8388608
    __bf16* Xp    = (__bf16*)(ws + 142606336);         // 8192*256*2    =   4194304
    __bf16* BmT   = (__bf16*)(ws + 146800640);         // 8192*256*2    =   4194304
    float*  Bbias = (float*) (ws + 150994944);         // 8192*64*4     =   2097152
    float*  xroot = (float*) (ws + 153092096);         // 8192*64*4     =   2097152
    float*  agg   = (float*) (ws + 155189248);         // 8192*64*4     =   2097152
    float*  psum  = (float*) (ws + 157286400);         // 512*64*4      =    131072
    float*  cnt   = (float*) (ws + 157417472);         // 512*4         =      2048
    // total: 157419520 bytes (~150 MB)

    // zero the accumulation buffers (agg, psum, cnt are contiguous)
    hipMemsetAsync(agg, 0, 2097152 + 131072 + 2048, stream);

    // stage 1: independent prep kernels
    edge_mlp1  <<<EE / 2, 256, 0, stream>>>(eattr, W1, b1, he);
    pack_x     <<<NN,     256, 0, stream>>>(x, Xp);
    pack_w2    <<<JD,     256, 0, stream>>>(W2, BmT);
    node_linear<<<NN / 4, 256, 0, stream>>>(x, b2, root, conv_b, Bbias, xroot);

    // stage 2: the WMMA GEMM (compute-bound core, TDM-staged B panel)
    dim3 ggrid(JD / 64, NN / 128);
    gemm_wmma<<<ggrid, 256, 0, stream>>>(Xp, BmT, U);

    // stage 3: per-edge messages + scatter-add
    edge_msg<<<EE / 4, 256, 0, stream>>>(U, he, Bbias, ei, agg);

    // stage 4: NNConv activation + mean-pool accumulation
    relu_pool<<<(NN * HH) / 256, 256, 0, stream>>>(agg, xroot, batch, psum, cnt);

    // stage 5: MLP head
    head_mlp<<<GG, 256, 0, stream>>>(psum, cnt, fw1, fb1, fw2, fb2, fw3, fb3,
                                     fw4, fb4, out);
}
